// ReturnSpecializedBlock_78176994722018
// MI455X (gfx1250) — compile-verified, matrix operand order checked by reading, not run on hardware
//
#include <hip/hip_runtime.h>
#include <hip/hip_bf16.h>

// ---------------------------------------------------------------------------
// Model constants
// ---------------------------------------------------------------------------
#define Dm   512
#define Sm   64
#define DFF  2048
#define Bn   8
#define Lm   2048
#define Mr   (Bn * Lm)        // 16384 rows
#define HIDN 128              // fuse gate hidden (D/4)
#define CH   32               // scan chunks
#define CLch ((Lm + CH - 1) / CH)

// ---------------------------------------------------------------------------
// WMMA GEMM: C = act((acc?C:0) + A*W + bias) (+ resid | resid*sigmoid)
// A: [M,K] f32 (row stride ldA, optional causal row-shift inside length-Lm
// sequences).  W: [K,N] f32 row-major.  f16 WMMA 16x16x32, f32 accumulate.
// BM=BN=128, BK=64 (two WMMA k-steps per barrier epoch), 8 waves/block.
// ---------------------------------------------------------------------------
#define BM 128
#define BN 128
#define BK 64
#define LSTR (BK + 8)   // LDS row stride in halves: 72 (=144B, 16B aligned)

typedef _Float16 v16h __attribute__((ext_vector_type(16)));
typedef _Float16 v8h  __attribute__((ext_vector_type(8)));
typedef _Float16 v4h  __attribute__((ext_vector_type(4)));
typedef float    v8f  __attribute__((ext_vector_type(8)));

__device__ __forceinline__ float gelu_f(float v) {
  return 0.5f * v * (1.0f + erff(v * 0.70710678118654752f));
}
__device__ __forceinline__ float sigm_f(float v) {
  return 1.0f / (1.0f + __expf(-v));
}

// Load one 16x32(K) f16 fragment (ISA 7.12.2 layout): lane-half h reads two
// contiguous 16B chunks of its row -> 2x ds_load_b128.
__device__ __forceinline__ v16h load_frag(const _Float16* rowp, int kk, int h) {
  const v8h lo = *(const v8h*)(rowp + kk + h * 8);
  const v8h hi = *(const v8h*)(rowp + kk + 16 + h * 8);
  return __builtin_shufflevector(lo, hi, 0, 1, 2, 3, 4, 5, 6, 7,
                                 8, 9, 10, 11, 12, 13, 14, 15);
}

// ACT: 0=none 1=gelu 2=sigmoid 3=tanh 4=(out = resid * sigmoid(v))
template <int ACT, bool ACCF, bool BIASF, bool RESF>
__global__ __launch_bounds__(256)
void gemm_wmma(const float* __restrict__ A, long ldA,
               const float* __restrict__ W,
               const float* __restrict__ bias,
               const float* __restrict__ resid, long ldR,
               float* __restrict__ C, long ldC,
               int M, int N, int K, int shift) {
  __shared__ _Float16 As[BM][LSTR];
  __shared__ _Float16 Bs[BN][LSTR];

  const int tid   = threadIdx.x;
  const int lane  = tid & 31;
  const int wave  = tid >> 5;
  const int wm0   = (wave >> 1) * 32;   // 4 waves along M
  const int wn0   = (wave & 1) * 64;    // 2 waves along N
  const int m0    = blockIdx.y * BM;
  const int n0    = blockIdx.x * BN;
  const int half_ = lane >> 4;
  const int r16   = lane & 15;

  v8f acc[2][4];
#pragma unroll
  for (int i = 0; i < 2; ++i)
#pragma unroll
    for (int j = 0; j < 4; ++j)
#pragma unroll
      for (int e = 0; e < 8; ++e) acc[i][j][e] = 0.0f;

  // A staging: 128 rows x 64 cols, 2 threads/row, 32 floats (8 float4) each
  const int atr = tid >> 1, atc = (tid & 1) * 32;
  // W staging: per-thread column n_loc, k-group of 32
  const int wn = tid & 127, wkg = (tid >> 7) * 32;

  // A row source (causal shift within length-Lm sequences), computed once
  const int  gr    = m0 + atr;
  bool       rowok = gr < M;
  long       srow  = gr;
  if (shift > 0 && rowok) {
    int l = gr % Lm, ls = l - shift;
    if (ls < 0) rowok = false;
    else        srow = (long)(gr / Lm) * Lm + ls;
  }
  const float* arow = A + srow * ldA;

  for (int k0 = 0; k0 < K; k0 += BK) {
    // ---- prefetch next W K-panel (global_prefetch_b8) ----
    if (k0 + BK < K && (n0 + wn) < N)
      __builtin_prefetch(W + (long)(k0 + BK + wkg) * N + n0 + wn, 0, 1);

    // ---- stage A tile (f32 -> f16, packed v4h stores) ----
#pragma unroll
    for (int q = 0; q < 8; ++q) {
      int c = atc + q * 4;
      float4 v = make_float4(0.f, 0.f, 0.f, 0.f);
      if (rowok && (k0 + c) < K) v = *(const float4*)(arow + k0 + c);
      v4h hv;
      hv[0] = (_Float16)v.x; hv[1] = (_Float16)v.y;
      hv[2] = (_Float16)v.z; hv[3] = (_Float16)v.w;
      *(v4h*)&As[atr][c] = hv;
    }
    // ---- stage W tile transposed: column reads (coalesced across lanes),
    //      vectorized ds_store_b128 of Bs[n][k] ----
    {
      bool nok = (n0 + wn) < N;
      const float* wcol = W + n0 + wn;
#pragma unroll
      for (int qq = 0; qq < 4; ++qq) {
        v8h hv;
#pragma unroll
        for (int e = 0; e < 8; ++e) {
          int k = wkg + qq * 8 + e;
          float wv = (nok && (k0 + k) < K) ? wcol[(long)(k0 + k) * N] : 0.f;
          hv[e] = (_Float16)wv;
        }
        *(v8h*)&Bs[wn][wkg + qq * 8] = hv;
      }
    }
    __syncthreads();

    // ---- two WMMA k-steps per epoch ----
#pragma unroll
    for (int kk = 0; kk < BK; kk += 32) {
      v16h bfrag[4];
#pragma unroll
      for (int j = 0; j < 4; ++j)
        bfrag[j] = load_frag(&Bs[wn0 + j * 16 + r16][0], kk, half_);
#pragma unroll
      for (int i = 0; i < 2; ++i) {
        v16h afrag = load_frag(&As[wm0 + i * 16 + r16][0], kk, half_);
#pragma unroll
        for (int j = 0; j < 4; ++j) {
          acc[i][j] = __builtin_amdgcn_wmma_f32_16x16x32_f16(
              false, afrag, false, bfrag[j], (short)0, acc[i][j], false, false);
        }
      }
    }
    __syncthreads();
  }

  // ---- epilogue: C/D layout (VGPR e: M=e+8*half, N=lane&15), no branches ----
#pragma unroll
  for (int i = 0; i < 2; ++i)
#pragma unroll
    for (int j = 0; j < 4; ++j)
#pragma unroll
      for (int e = 0; e < 8; ++e) {
        int row = m0 + wm0 + i * 16 + e + half_ * 8;
        int col = n0 + wn0 + j * 16 + r16;
        if (row < M && col < N) {
          float v = acc[i][j][e];
          long cidx = (long)row * ldC + col;
          if constexpr (ACCF)  v += C[cidx];
          if constexpr (BIASF) v += bias[col];
          if constexpr (ACT == 1) v = gelu_f(v);
          else if constexpr (ACT == 2) v = sigm_f(v);
          else if constexpr (ACT == 3) v = tanhf(v);
          if constexpr (RESF) {
            float rv = resid[(long)row * ldR + col];
            if constexpr (ACT == 4) v = rv * sigm_f(v);
            else                    v += rv;
          }
          C[cidx] = v;
        }
      }
}

// ---------------------------------------------------------------------------
// RMSNorm: out = (x [+ add]) * rsqrt(mean((x+add)^2)+eps) * w   (one row/block)
// ---------------------------------------------------------------------------
__global__ __launch_bounds__(256)
void rms_kernel(const float* __restrict__ x, const float* __restrict__ addl,
                const float* __restrict__ w, float* __restrict__ out, int Dd) {
  long row = blockIdx.x;
  const float* px = x + row * Dd;
  const float* pa = addl ? addl + row * Dd : nullptr;
  __shared__ float sred[256];
  float ss = 0.f;
  for (int d = threadIdx.x; d < Dd; d += 256) {
    float v = px[d] + (pa ? pa[d] : 0.f);
    ss += v * v;
  }
  sred[threadIdx.x] = ss;
  __syncthreads();
  for (int o = 128; o > 0; o >>= 1) {
    if (threadIdx.x < o) sred[threadIdx.x] += sred[threadIdx.x + o];
    __syncthreads();
  }
  float scale = rsqrtf(sred[0] / Dd + 1e-8f);
  for (int d = threadIdx.x; d < Dd; d += 256) {
    float v = px[d] + (pa ? pa[d] : 0.f);
    out[row * Dd + d] = v * scale * w[d];
  }
}

// ---------------------------------------------------------------------------
// Patch mean-pool: tok[b,n,d] = mean_j x[b, n*stride+j, d]  (zero-pad past L)
// ---------------------------------------------------------------------------
__global__ void patch_pool(const float* __restrict__ x, float* __restrict__ tok,
                           int Bx, int Lx, int Dx, int Np, int plen, int stride) {
  long idx = (long)blockIdx.x * blockDim.x + threadIdx.x;
  long tot = (long)Bx * Np * Dx;
  if (idx >= tot) return;
  int d = idx % Dx;
  long t = idx / Dx;
  int n = t % Np, b = t / Np;
  float s = 0.f;
  for (int j = 0; j < plen; ++j) {
    int l = n * stride + j;
    if (l < Lx) s += x[((long)b * Lx + l) * Dx + d];
  }
  tok[idx] = s / (float)plen;
}

// Linear interp (align_corners=False) to L_out, writing a column slice.
__global__ void interp_kernel(const float* __restrict__ tok, float* __restrict__ out,
                              long ldOut, int Bx, int Lout, int Dx, int Np) {
  long idx = (long)blockIdx.x * blockDim.x + threadIdx.x;
  long tot = (long)Bx * Lout * Dx;
  if (idx >= tot) return;
  int d = idx % Dx;
  long t = idx / Dx;
  int l = t % Lout, b = t / Lout;
  float src = ((l + 0.5f) * (float)Np) / (float)Lout - 0.5f;
  src = fminf(fmaxf(src, 0.f), (float)Np - 1.f);
  int i0 = (int)floorf(src);
  int i1 = (i0 + 1 < Np) ? i0 + 1 : Np - 1;
  float wgt = src - (float)i0;
  float v = tok[((long)b * Np + i0) * Dx + d] * (1.f - wgt) +
            tok[((long)b * Np + i1) * Dx + d] * wgt;
  out[((long)b * Lout + l) * ldOut + d] = v;
}

// Strided column-block copy (for concat assembly)
__global__ void copy_cols(const float* __restrict__ src, long ldS,
                          float* __restrict__ dst, long ldD, long Mrows, int ncols) {
  long idx = (long)blockIdx.x * blockDim.x + threadIdx.x;
  long tot = Mrows * ncols;
  if (idx >= tot) return;
  long r = idx / ncols;
  int  c = idx % ncols;
  dst[r * ldD + c] = src[r * ldS + c];
}

// ---------------------------------------------------------------------------
// Chunked parallel scan for h[t] = (1-d)h + d*u ; y = h*sigmoid(g)
// ---------------------------------------------------------------------------
__global__ void scan_p1(const float* __restrict__ ut, const float* __restrict__ dt,
                        float* __restrict__ chA, float* __restrict__ chB,
                        int Bx, int Lx, int Sx) {
  int idx = blockIdx.x * blockDim.x + threadIdx.x;
  int tot = Bx * Sx * CH;
  if (idx >= tot) return;
  int c = idx % CH;
  int bs = idx / CH;
  int s = bs % Sx, b = bs / Sx;
  long base = (long)b * Lx * Sx + s;
  float a = 1.f, acc = 0.f;
  int t0 = c * CLch, t1 = (t0 + CLch < Lx) ? t0 + CLch : Lx;
  for (int t = t0; t < t1; ++t) {
    float d = dt[base + (long)t * Sx];
    float u = ut[base + (long)t * Sx];
    float om = 1.f - d;
    a *= om;
    acc = om * acc + d * u;
  }
  chA[idx] = a;
  chB[idx] = acc;
}

__global__ void scan_p2(const float* __restrict__ chA, const float* __restrict__ chB,
                        float* __restrict__ hst, int totBS) {
  int idx = blockIdx.x * blockDim.x + threadIdx.x;
  if (idx >= totBS) return;
  float h = 0.f;
  for (int c = 0; c < CH; ++c) {
    hst[idx * CH + c] = h;
    h = chA[idx * CH + c] * h + chB[idx * CH + c];
  }
}

__global__ void scan_p3(const float* __restrict__ ut, const float* __restrict__ dt,
                        const float* __restrict__ inp, const float* __restrict__ hst,
                        float* __restrict__ y, int Bx, int Lx, int Sx) {
  int idx = blockIdx.x * blockDim.x + threadIdx.x;
  int tot = Bx * Sx * CH;
  if (idx >= tot) return;
  int c = idx % CH;
  int bs = idx / CH;
  int s = bs % Sx, b = bs / Sx;
  long base = (long)b * Lx * Sx + s;
  float h = hst[idx];
  int t0 = c * CLch, t1 = (t0 + CLch < Lx) ? t0 + CLch : Lx;
  for (int t = t0; t < t1; ++t) {
    float d = dt[base + (long)t * Sx];
    float u = ut[base + (long)t * Sx];
    h = (1.f - d) * h + d * u;
    float g = inp[((long)b * Lx + t) * (3 * Sm) + 2 * Sm + s];
    y[base + (long)t * Sx] = h * sigm_f(g);
  }
}

// ---------------------------------------------------------------------------
// Workspace arena (float offsets)
// ---------------------------------------------------------------------------
#define MD        ((size_t)Mr * Dm)            // 8,388,608
#define NPMAX     511
#define OFF_XB    ((size_t)0)
#define OFF_NB    (OFF_XB  + MD)
#define OFF_HB    (OFF_NB  + MD)
#define OFF_OB    (OFF_HB  + MD)
#define OFF_XL    (OFF_OB  + MD)
#define OFF_HID   (OFF_XL  + MD)               // Mr*DFF
#define OFF_CAT   (OFF_HID + (size_t)Mr * DFF) // Mr*4D
#define OFF_C3    (OFF_CAT + (size_t)Mr * 4 * Dm)
#define OFF_INP   (OFF_C3  + (size_t)Mr * 3 * Dm)
#define OFF_UT    (OFF_INP + (size_t)Mr * 3 * Sm)
#define OFF_DT    (OFF_UT  + (size_t)Mr * Sm)
#define OFF_YB    (OFF_DT  + (size_t)Mr * Sm)
#define OFF_TOK   (OFF_YB  + (size_t)Mr * Sm)
#define OFF_TKN   (OFF_TOK + (size_t)Bn * NPMAX * Dm)
#define OFF_TKH   (OFF_TKN + (size_t)Bn * NPMAX * Dm)
#define OFF_G1    (OFF_TKH + (size_t)Bn * NPMAX * DFF)
#define OFF_SC    (OFF_G1  + (size_t)Mr * HIDN)

// Dispatch over the 9 (ACT, ACC, BIAS, RES) combos this model uses.
static void launch_gemm(hipStream_t st, const float* A, long ldA, const float* W,
                        const float* bias, const float* resid, long ldR,
                        float* C, long ldC, int M, int N, int K, int shift,
                        bool accf, int act) {
  dim3 g((N + BN - 1) / BN, (M + BM - 1) / BM);
  const bool hb = bias != nullptr, hr = resid != nullptr;
#define GCALL(ACTV, ACCV, BIASV, RESV)                                         \
  gemm_wmma<ACTV, ACCV, BIASV, RESV><<<g, 256, 0, st>>>(                       \
      A, ldA, W, bias, resid, ldR, C, ldC, M, N, K, shift)
  if      (!accf &&  hb && !hr && act == 0) GCALL(0, false, true,  false);
  else if ( accf && !hb && !hr && act == 0) GCALL(0, true,  false, false);
  else if ( accf && !hb && !hr && act == 1) GCALL(1, true,  false, false);
  else if ( accf && !hb &&  hr && act == 0) GCALL(0, true,  false, true );
  else if (!accf &&  hb && !hr && act == 1) GCALL(1, false, true,  false);
  else if (!accf &&  hb &&  hr && act == 0) GCALL(0, false, true,  true );
  else if (!accf &&  hb && !hr && act == 3) GCALL(3, false, true,  false);
  else if (!accf &&  hb && !hr && act == 2) GCALL(2, false, true,  false);
  else if (!accf &&  hb &&  hr && act == 4) GCALL(4, false, true,  true );
  else                                      GCALL(0, false, true,  false);
#undef GCALL
}

extern "C" void kernel_launch(void* const* d_in, const int* in_sizes, int n_in,
                              void* d_out, int out_size, void* d_ws, size_t ws_size,
                              hipStream_t stream) {
  (void)in_sizes; (void)n_in; (void)out_size; (void)ws_size;
  const float* X   = (const float*)d_in[0];
  const float* VOL = (const float*)d_in[1];
  auto P = [&](int i) { return (const float*)d_in[i]; };
  // jax tree-flatten (sorted dict keys), params start at index 2:
  //  2..5  final_ffn{b1,b2,w1,w2}; 6 final_norm
  //  7..15 fuse{fb1,fb2,fw1,fw2,gb1,gb2,gw1,gw2,norm}
  // 16..24 local0{c1b,c1w,c2b,c2w,ffn.b1,ffn.b2,ffn.w1,ffn.w2,norm}; 25..33 local1
  // 34..47 long0{ffn.b1,b2,w1,w2, m.dt_b,dt_w,in_b,in_w,norm,out_b,out_w,x_b,x_w, norm}; 48..61 long1
  // 62..73 patch0{m1.b1,b2,w1,w2, m2.b1,b2,w1,w2, n1,n2,proj_b,proj_w}; 74..85 p1; 86..97 p2
  // 98 patch_out_b; 99 patch_out_w

  float* ws  = (float*)d_ws;
  float* XB  = ws + OFF_XB;   float* NB  = ws + OFF_NB;
  float* HB  = ws + OFF_HB;   float* OB  = ws + OFF_OB;
  float* XL  = ws + OFF_XL;   float* HID = ws + OFF_HID;
  float* CAT = ws + OFF_CAT;  float* C3  = ws + OFF_C3;
  float* INP = ws + OFF_INP;  float* UT  = ws + OFF_UT;
  float* DTb = ws + OFF_DT;   float* YB  = ws + OFF_YB;
  float* TOK = ws + OFF_TOK;  float* TKN = ws + OFF_TKN;
  float* TKH = ws + OFF_TKH;  float* G1  = ws + OFF_G1;
  float* chA = ws + OFF_SC;
  float* chB = chA + (size_t)Bn * Sm * CH;
  float* HST = chB + (size_t)Bn * Sm * CH;

  auto gemm = [&](const float* Aa, long lda, const float* Ww, const float* bb,
                  const float* rr, long ldr, float* Cc, long ldc,
                  int Mm, int Nn, int Kk, int sh, int accf, int act) {
    launch_gemm(stream, Aa, lda, Ww, bb, rr, ldr, Cc, ldc, Mm, Nn, Kk, sh,
                accf != 0, act);
  };
  auto rms = [&](const float* in1, const float* add, const float* w, float* out, int rows) {
    rms_kernel<<<rows, 256, 0, stream>>>(in1, add, w, out, Dm);
  };
  auto nblk = [](long n) { return (unsigned)((n + 255) / 256); };

  hipMemcpyAsync(XB, X, MD * sizeof(float), hipMemcpyDeviceToDevice, stream);
  hipMemcpyAsync(XL, X, MD * sizeof(float), hipMemcpyDeviceToDevice, stream);

  // ---------------- Local shock branch (2 layers, dil 1,2) ----------------
  for (int li = 0; li < 2; ++li) {
    int base = 16 + li * 9;
    const float *c1b = P(base+0), *c1w = P(base+1), *c2b = P(base+2), *c2w = P(base+3);
    const float *fb1 = P(base+4), *fb2 = P(base+5), *fw1 = P(base+6), *fw2 = P(base+7);
    const float *nw  = P(base+8);
    int dil = 1 << li;
    rms(XB, nullptr, nw, NB, Mr);
    for (int k = 0; k < 3; ++k)    // conv1 + GELU
      gemm(NB, Dm, c1w + (size_t)k * Dm * Dm, (k == 0) ? c1b : nullptr,
           nullptr, 0, HB, Dm, Mr, Dm, Dm, (2 - k) * dil, k > 0, (k == 2) ? 1 : 0);
    for (int k = 0; k < 3; ++k)    // conv2, last tap adds residual x -> NB
      gemm(HB, Dm, c2w + (size_t)k * Dm * Dm, (k == 0) ? c2b : nullptr,
           (k == 2) ? XB : nullptr, Dm, NB, Dm, Mr, Dm, Dm, (2 - k) * dil, k > 0, 0);
    rms(NB, nullptr, nw, HB, Mr);  // same norm weight reused (per reference)
    gemm(HB, Dm, fw1, fb1, nullptr, 0, HID, DFF, Mr, DFF, Dm, 0, 0, 1);
    gemm(HID, DFF, fw2, fb2, NB, Dm, XB, Dm, Mr, Dm, DFF, 0, 0, 0);
  }

  // ---------------- Patch branch (plens 8,16,32) ----------------
  const int plens[3] = {8, 16, 32};
  for (int pi = 0; pi < 3; ++pi) {
    int pb = 62 + pi * 12;
    const float *m1b1 = P(pb+0), *m1b2 = P(pb+1), *m1w1 = P(pb+2), *m1w2 = P(pb+3);
    const float *m2b1 = P(pb+4), *m2b2 = P(pb+5), *m2w1 = P(pb+6), *m2w2 = P(pb+7);
    const float *n1 = P(pb+8), *n2 = P(pb+9), *projb = P(pb+10), *projw = P(pb+11);
    int plen = plens[pi], stride = plen / 2;
    int rem = (Lm - plen) % stride, pad = rem ? stride - rem : 0;
    int Np = (Lm + pad - plen) / stride + 1;
    int Mp = Bn * Np;
    patch_pool<<<nblk((long)Mp * Dm), 256, 0, stream>>>(X, TOK, Bn, Lm, Dm, Np, plen, stride);
    gemm(TOK, Dm, projw, projb, nullptr, 0, TKN, Dm, Mp, Dm, Dm, 0, 0, 0);
    rms(TKN, nullptr, n1, TOK, Mp);
    gemm(TOK, Dm, m1w1, m1b1, nullptr, 0, TKH, DFF, Mp, DFF, Dm, 0, 0, 1);
    gemm(TKH, DFF, m1w2, m1b2, TKN, Dm, TKN, Dm, Mp, Dm, DFF, 0, 0, 0);
    rms(TKN, nullptr, n2, TOK, Mp);
    gemm(TOK, Dm, m2w1, m2b1, nullptr, 0, TKH, DFF, Mp, DFF, Dm, 0, 0, 1);
    gemm(TKH, DFF, m2w2, m2b2, TKN, Dm, TKN, Dm, Mp, Dm, DFF, 0, 0, 0);
    interp_kernel<<<nblk((long)Mr * Dm), 256, 0, stream>>>(TKN, C3 + pi * Dm, 3 * Dm, Bn, Lm, Dm, Np);
  }
  // patch_out projection straight into concat column D
  gemm(C3, 3 * Dm, P(99), P(98), nullptr, 0, CAT + Dm, 4 * Dm, Mr, Dm, 3 * Dm, 0, 0, 0);

  // ---------------- Long memory branch (2 layers) ----------------
  for (int li = 0; li < 2; ++li) {
    int lb = 34 + li * 14;
    const float *fb1 = P(lb+0), *fb2 = P(lb+1), *fw1 = P(lb+2), *fw2 = P(lb+3);
    const float *dtbi = P(lb+4), *dtw = P(lb+5), *inb = P(lb+6), *inw = P(lb+7);
    const float *mn = P(lb+8), *outb = P(lb+9), *outw = P(lb+10);
    const float *xbi = P(lb+11), *xw = P(lb+12), *nw = P(lb+13);
    rms(XL, nullptr, mn, NB, Mr);
    gemm(NB, Dm, inw, inb, nullptr, 0, INP, 3 * Sm, Mr, 3 * Sm, Dm, 0, 0, 0);
    gemm(INP,      3 * Sm, xw,  xbi,  nullptr, 0, UT,  Sm, Mr, Sm, Sm, 0, 0, 3); // tanh
    gemm(INP + Sm, 3 * Sm, dtw, dtbi, nullptr, 0, DTb, Sm, Mr, Sm, Sm, 0, 0, 2); // sigmoid
    scan_p1<<<nblk(Bn * Sm * CH), 256, 0, stream>>>(UT, DTb, chA, chB, Bn, Lm, Sm);
    scan_p2<<<nblk(Bn * Sm), 256, 0, stream>>>(chA, chB, HST, Bn * Sm);
    scan_p3<<<nblk(Bn * Sm * CH), 256, 0, stream>>>(UT, DTb, INP, HST, YB, Bn, Lm, Sm);
    gemm(YB, Sm, outw, outb, XL, Dm, XL, Dm, Mr, Dm, Sm, 0, 0, 0);
    rms(XL, nullptr, nw, NB, Mr);
    gemm(NB, Dm, fw1, fb1, nullptr, 0, HID, DFF, Mr, DFF, Dm, 0, 0, 1);
    gemm(HID, DFF, fw2, fb2, XL, Dm, XL, Dm, Mr, Dm, DFF, 0, 0, 0);
  }

  // ---------------- Concat [xl, xp, xlong, vol] ----------------
  copy_cols<<<nblk(MD), 256, 0, stream>>>(XB,  Dm, CAT,           4 * Dm, Mr, Dm);
  copy_cols<<<nblk(MD), 256, 0, stream>>>(XL,  Dm, CAT + 2 * Dm,  4 * Dm, Mr, Dm);
  copy_cols<<<nblk(MD), 256, 0, stream>>>(VOL, Dm, CAT + 3 * Dm,  4 * Dm, Mr, Dm);

  // ---------------- Fuse ----------------
  gemm(CAT, 4 * Dm, P(13), P(11), nullptr, 0, G1, HIDN, Mr, HIDN, 4 * Dm, 0, 0, 1); // gelu
  gemm(G1, HIDN, P(14), P(12), CAT, 4 * Dm, CAT, 4 * Dm, Mr, 4 * Dm, HIDN, 0, 0, 4); // cat *= sigmoid
  gemm(CAT, 4 * Dm, P(9), P(7), nullptr, 0, HB, Dm, Mr, Dm, 4 * Dm, 0, 0, 1);        // gelu
  gemm(HB, Dm, P(10), P(8), nullptr, 0, NB, Dm, Mr, Dm, Dm, 0, 0, 0);                 // fz
  rms(X, NB, P(15), OB, Mr);          // out = rms(x + fz, fuse.norm)

  // ---------------- Final FFN ----------------
  rms(OB, nullptr, P(6), NB, Mr);
  gemm(NB, Dm, P(4), P(2), nullptr, 0, HID, DFF, Mr, DFF, Dm, 0, 0, 1);
  gemm(HID, DFF, P(5), P(3), OB, Dm, (float*)d_out, Dm, Mr, Dm, DFF, 0, 0, 0);
}
